// CausalAttentionBlock_5093831213186
// MI455X (gfx1250) — matrix operand order repaired and checked by reading, not compiled, not run on hardware
//
#include <hip/hip_runtime.h>

// ============================================================================
// CausalAttentionBlock for MI455X (gfx1250, wave32, WMMA).
// All GEMMs run on v_wmma_f32_16x16x32_bf16 (bf16 in, f32 accumulate).
// Global->LDS staging uses CDNA5 async-LDS copies + s_wait_asynccnt.
// ============================================================================

typedef __bf16 bf16;
typedef __attribute__((ext_vector_type(16))) __bf16 bf16x16;
typedef __attribute__((ext_vector_type(8)))  float   f32x8;
typedef int i32x4 __attribute__((vector_size(16)));

#define DEVFN static __device__ __forceinline__

// Problem dims: B=2, S=2048, H=1024, NH=16, HD=64
constexpr int Bb = 2, Ss = 2048, Hh = 1024, NH = 16, HD = 64;
constexpr int Mrows = Bb * Ss;  // 4096 token rows

// ---------------- CDNA5 async global->LDS copy (guarded) --------------------
#if defined(__has_builtin)
#  if __has_builtin(__builtin_amdgcn_global_load_async_to_lds_b128)
#    define HAVE_ASYNC_LDS 1
#  endif
#  if __has_builtin(__builtin_amdgcn_s_wait_asynccnt)
#    define HAVE_WAIT_ASYNC 1
#  endif
#endif

DEVFN void cp16(void* lds_dst, const void* gsrc) {
#if defined(HAVE_ASYNC_LDS)
  __builtin_amdgcn_global_load_async_to_lds_b128(
      (__attribute__((address_space(1))) i32x4*)(gsrc),
      (__attribute__((address_space(3))) i32x4*)(lds_dst), 0, 0);
#else
  *(uint4*)lds_dst = *(const uint4*)gsrc;
#endif
}

DEVFN void wait_cp() {
#if defined(HAVE_ASYNC_LDS)
#  if defined(HAVE_WAIT_ASYNC)
  __builtin_amdgcn_s_wait_asynccnt(0);
#  else
  asm volatile("s_wait_asynccnt 0x0" ::: "memory");
#  endif
#endif
}

DEVFN f32x8 zero8() { f32x8 z = {0.f,0.f,0.f,0.f,0.f,0.f,0.f,0.f}; return z; }

DEVFN f32x8 wmma_bf16(bf16x16 a, bf16x16 b, f32x8 c) {
  // 8 args: (neg_a, A, neg_b, B, c_mod, C, reuse_a, reuse_b)
  return __builtin_amdgcn_wmma_f32_16x16x32_bf16(false, a, false, b, (short)0, c,
                                                 false, false);
}

union Frag { uint4 u[2]; bf16x16 v; };

// ---------------- weight convert fp32[K][N] -> bf16[N][K] -------------------
__global__ __launch_bounds__(256) void wconv_kernel(const float* __restrict__ W,
                                                    bf16* __restrict__ Wt,
                                                    int K, int N) {
  size_t i = (size_t)blockIdx.x * 256 + threadIdx.x;
  if (i >= (size_t)K * N) return;
  int n = (int)(i % (size_t)N);
  int k = (int)(i / (size_t)N);
  Wt[(size_t)n * K + k] = (bf16)W[i];
}

// ---------------- layernorm (fp32 in) -> bf16 out ---------------------------
// one wave per row; 8 rows per 256-thread block
__global__ __launch_bounds__(256) void ln_kernel(const float* __restrict__ x,
                                                 const float* __restrict__ gam,
                                                 const float* __restrict__ bet,
                                                 bf16* __restrict__ o, int nrows) {
  const int wave = threadIdx.x >> 5, lane = threadIdx.x & 31;
  const int row = blockIdx.x * 8 + wave;
  if (row >= nrows) return;
  const float4* xr = (const float4*)(x + (size_t)row * Hh);
  float4 v[8];
  float s = 0.f, sq = 0.f;
#pragma unroll
  for (int i = 0; i < 8; ++i) {
    v[i] = xr[i * 32 + lane];
    s  += v[i].x + v[i].y + v[i].z + v[i].w;
    sq += v[i].x*v[i].x + v[i].y*v[i].y + v[i].z*v[i].z + v[i].w*v[i].w;
  }
#pragma unroll
  for (int m = 1; m < 32; m <<= 1) {
    s  += __shfl_xor(s, m, 32);
    sq += __shfl_xor(sq, m, 32);
  }
  const float mu   = s * (1.f / Hh);
  const float rstd = rsqrtf(sq * (1.f / Hh) - mu * mu + 1e-12f);
  const float4* g4 = (const float4*)gam;
  const float4* b4 = (const float4*)bet;
  bf16* orow = o + (size_t)row * Hh;
#pragma unroll
  for (int i = 0; i < 8; ++i) {
    float4 g = g4[i * 32 + lane], b = b4[i * 32 + lane];
    int c = (i * 32 + lane) * 4;
    orow[c + 0] = (bf16)((v[i].x - mu) * rstd * g.x + b.x);
    orow[c + 1] = (bf16)((v[i].y - mu) * rstd * g.y + b.y);
    orow[c + 2] = (bf16)((v[i].z - mu) * rstd * g.z + b.z);
    orow[c + 3] = (bf16)((v[i].w - mu) * rstd * g.w + b.w);
  }
}

// ---------------- WMMA GEMM: C = A[M][K] * Bt[N][K]^T -----------------------
constexpr int BM = 128, BN = 128, BK = 32;
constexpr int LDT = BK + 8;  // padded LDS row (bf16 elems) -> conflict-free frags

// EPI 0: +bias, scatter bf16 into per-head q/k/v [B][NH][S][HD]
// EPI 1: +bias +resid(f32) -> f32 out
// EPI 2: exact GELU -> bf16 out
template <int EPI>
__global__ __launch_bounds__(256) void gemm_kernel(
    const bf16* __restrict__ A, const bf16* __restrict__ Bt,
    const float* __restrict__ bias, const float* __restrict__ resid,
    float* __restrict__ outF, bf16* __restrict__ outB,
    bf16* __restrict__ qO, bf16* __restrict__ kO, bf16* __restrict__ vO,
    int M, int N, int K) {
  __shared__ bf16 sA[2][BM][LDT];
  __shared__ bf16 sB[2][BN][LDT];

  const int t = threadIdx.x;
  const int wave = t >> 5, lane = t & 31;
  const int lrow = lane & 15, lhalf = lane >> 4;
  const int mbase = blockIdx.y * BM, nbase = blockIdx.x * BN;
  const int srow = t >> 1, sseg = t & 1;  // staging: 2 threads per 64B tile row
  const int nk = K / BK;

  f32x8 acc[8];
#pragma unroll
  for (int i = 0; i < 8; ++i) acc[i] = zero8();

  auto stage = [&](int kt, int bufI) {
    const bf16* ga = A  + (size_t)(mbase + srow) * K + kt * BK + sseg * 16;
    const bf16* gb = Bt + (size_t)(nbase + srow) * K + kt * BK + sseg * 16;
    cp16(&sA[bufI][srow][sseg * 16],     ga);
    cp16(&sA[bufI][srow][sseg * 16 + 8], ga + 8);
    cp16(&sB[bufI][srow][sseg * 16],     gb);
    cp16(&sB[bufI][srow][sseg * 16 + 8], gb + 8);
  };

  int buf = 0;
  stage(0, 0);
  wait_cp();
  __syncthreads();

  for (int kt = 0; kt < nk; ++kt) {
    if (kt + 1 < nk) stage(kt + 1, buf ^ 1);  // prefetch next tile
    // A fragment: row = lane&15 (of this wave's strip); K = e + 16*lhalf
    Frag af;
    const bf16* ar = &sA[buf][wave * 16 + lrow][0];
    af.u[0] = *(const uint4*)(ar + lhalf * 16);
    af.u[1] = *(const uint4*)(ar + lhalf * 16 + 8);
#pragma unroll
    for (int nt = 0; nt < 8; ++nt) {
      // B fragment: N = lane&15, K = e + 16*lhalf  (Bt rows are contiguous K)
      Frag bfr;
      const bf16* br = &sB[buf][nt * 16 + lrow][lhalf * 16];
      bfr.u[0] = *(const uint4*)(br);
      bfr.u[1] = *(const uint4*)(br + 8);
      acc[nt] = wmma_bf16(af.v, bfr.v, acc[nt]);
    }
    if (kt + 1 < nk) { wait_cp(); __syncthreads(); buf ^= 1; }
  }

  // epilogue: D element r -> row = r + 8*lhalf, col = lane&15
#pragma unroll
  for (int nt = 0; nt < 8; ++nt) {
#pragma unroll
    for (int r = 0; r < 8; ++r) {
      const int m = mbase + wave * 16 + r + 8 * lhalf;
      const int n = nbase + nt * 16 + lrow;
      float val = acc[nt][r];
      if constexpr (EPI == 0) {
        val += bias[n];
        const int bi = m >> 11, si = m & (Ss - 1);  // S = 2048
        bf16* dst; int nn = n;
        if (nn < Hh)          { dst = qO; }
        else if (nn < 2 * Hh) { dst = kO; nn -= Hh; }
        else                  { dst = vO; nn -= 2 * Hh; }
        const int hh = nn >> 6, dd = nn & 63;
        dst[(((size_t)bi * NH + hh) * Ss + si) * HD + dd] = (bf16)val;
      } else if constexpr (EPI == 1) {
        val += bias[n] + resid[(size_t)m * N + n];
        outF[(size_t)m * N + n] = val;
      } else {
        const float ge = 0.5f * val * (1.f + erff(val * 0.70710678f));
        outB[(size_t)m * N + n] = (bf16)ge;
      }
    }
  }
}

// ---------------- flash attention (causal, per-head) ------------------------
constexpr int QT = 128, KTile = 128;
constexpr int LK  = HD + 8;     // sK row pad
constexpr int LV  = KTile + 8;  // sVt row pad
constexpr int LPp = KTile + 8;  // sP row pad

__global__ __launch_bounds__(256) void attn_kernel(
    const bf16* __restrict__ Q, const bf16* __restrict__ Kmat,
    const bf16* __restrict__ Vmat, bf16* __restrict__ O /* [B*S][H] */) {
  __shared__ bf16 sK[KTile][LK];
  __shared__ bf16 sVt[HD][LV];
  __shared__ bf16 sP[8][16][LPp];  // per-wave P tile (16 q-rows x 128 keys)

  const int t = threadIdx.x;
  const int wave = t >> 5, lane = t & 31;
  const int lrow = lane & 15, lhalf = lane >> 4;
  const int qt = blockIdx.x, h = blockIdx.y, b = blockIdx.z;
  const size_t hb = ((size_t)b * NH + h) * Ss * HD;
  const int qbase = qt * QT;

  // Q fragments live in registers for the whole kernel (hd=64 -> 2 k-steps)
  Frag qf[2];
  {
    const bf16* qr = Q + hb + (size_t)(qbase + wave * 16 + lrow) * HD;
#pragma unroll
    for (int ks = 0; ks < 2; ++ks) {
      qf[ks].u[0] = *(const uint4*)(qr + ks * 32 + lhalf * 16);
      qf[ks].u[1] = *(const uint4*)(qr + ks * 32 + lhalf * 16 + 8);
    }
  }

  float statM[8], statL[8];
  f32x8 o[4];
#pragma unroll
  for (int r = 0; r < 8; ++r) { statM[r] = -1e30f; statL[r] = 0.f; }
#pragma unroll
  for (int i = 0; i < 4; ++i) o[i] = zero8();

  for (int kt = 0; kt <= qt; ++kt) {  // causal: only lower-triangular tiles
    const int kb = kt * KTile;
    {
      const int kr = t >> 1, seg = t & 1;
      const bf16* gk = Kmat + hb + (size_t)(kb + kr) * HD + seg * 32;
      cp16(&sK[kr][seg * 32],      gk);
      cp16(&sK[kr][seg * 32 + 8],  gk + 8);
      cp16(&sK[kr][seg * 32 + 16], gk + 16);
      cp16(&sK[kr][seg * 32 + 24], gk + 24);
      // V staged transposed: sVt[hd][key]
      const bf16* gv = Vmat + hb + (size_t)(kb + kr) * HD + seg * 32;
#pragma unroll
      for (int d = 0; d < 32; ++d) sVt[seg * 32 + d][kr] = gv[d];
    }
    wait_cp();
    __syncthreads();

    // S = Q K^T   (B-frag: N=key, K=hd, read from sK[key][hd])
    f32x8 sc[8];
#pragma unroll
    for (int nt = 0; nt < 8; ++nt) {
      f32x8 a = zero8();
#pragma unroll
      for (int ks = 0; ks < 2; ++ks) {
        Frag bfr;
        const bf16* br = &sK[nt * 16 + lrow][ks * 32 + lhalf * 16];
        bfr.u[0] = *(const uint4*)br;
        bfr.u[1] = *(const uint4*)(br + 8);
        a = wmma_bf16(qf[ks].v, bfr.v, a);
      }
      sc[nt] = a;
    }

    // scale + causal mask + running row max (rows = r + 8*lhalf)
    const bool diag = (kt == qt);
    float tmax[8];
#pragma unroll
    for (int r = 0; r < 8; ++r) tmax[r] = -1e30f;
#pragma unroll
    for (int nt = 0; nt < 8; ++nt) {
      const int key = kb + nt * 16 + lrow;
#pragma unroll
      for (int r = 0; r < 8; ++r) {
        float vv = sc[nt][r] * 0.125f;  // 1/sqrt(64)
        const int qrow = qbase + wave * 16 + r + 8 * lhalf;
        if (diag && key > qrow) vv = -1e30f;
        sc[nt][r] = vv;
        tmax[r] = fmaxf(tmax[r], vv);
      }
    }
#pragma unroll
    for (int r = 0; r < 8; ++r) {  // 16-lane reduce (wave32 halves are rows)
      float vv = tmax[r];
      vv = fmaxf(vv, __shfl_xor(vv, 1, 16));
      vv = fmaxf(vv, __shfl_xor(vv, 2, 16));
      vv = fmaxf(vv, __shfl_xor(vv, 4, 16));
      vv = fmaxf(vv, __shfl_xor(vv, 8, 16));
      tmax[r] = vv;
    }

    float rs[8];
#pragma unroll
    for (int r = 0; r < 8; ++r) {
      const float mnew = fmaxf(statM[r], tmax[r]);
      const float corr = __expf(statM[r] - mnew);
      statM[r] = mnew;
      statL[r] *= corr;
#pragma unroll
      for (int i = 0; i < 4; ++i) o[i][r] *= corr;
      rs[r] = 0.f;
    }
#pragma unroll
    for (int nt = 0; nt < 8; ++nt) {
#pragma unroll
      for (int r = 0; r < 8; ++r) {
        const float p = __expf(sc[nt][r] - statM[r]);
        rs[r] += p;
        sP[wave][r + 8 * lhalf][nt * 16 + lrow] = (bf16)p;  // reshape via LDS
      }
    }
#pragma unroll
    for (int r = 0; r < 8; ++r) {
      float vv = rs[r];
      vv += __shfl_xor(vv, 1, 16);
      vv += __shfl_xor(vv, 2, 16);
      vv += __shfl_xor(vv, 4, 16);
      vv += __shfl_xor(vv, 8, 16);
      statL[r] += vv;
    }

    // O += P V   (A-frag from sP, B-frag from sVt[hd][key])
#pragma unroll
    for (int ks = 0; ks < 4; ++ks) {
      Frag af;
      const bf16* pr = &sP[wave][lrow][ks * 32];
      af.u[0] = *(const uint4*)(pr + lhalf * 16);
      af.u[1] = *(const uint4*)(pr + lhalf * 16 + 8);
#pragma unroll
      for (int nt = 0; nt < 4; ++nt) {
        Frag bfr;
        const bf16* br = &sVt[nt * 16 + lrow][ks * 32 + lhalf * 16];
        bfr.u[0] = *(const uint4*)br;
        bfr.u[1] = *(const uint4*)(br + 8);
        o[nt] = wmma_bf16(af.v, bfr.v, o[nt]);
      }
    }
    __syncthreads();  // before restaging sK/sVt
  }

  // normalize + write attn output reassembled as [B*S][H] bf16
#pragma unroll
  for (int nt = 0; nt < 4; ++nt) {
#pragma unroll
    for (int r = 0; r < 8; ++r) {
      const int qrow = qbase + wave * 16 + r + 8 * lhalf;
      const int row  = b * Ss + qrow;
      O[(size_t)row * Hh + h * HD + nt * 16 + lrow] = (bf16)(o[nt][r] / statL[r]);
    }
  }
}

// ============================================================================
extern "C" void kernel_launch(void* const* d_in, const int* in_sizes, int n_in,
                              void* d_out, int out_size, void* d_ws, size_t ws_size,
                              hipStream_t stream) {
  (void)in_sizes; (void)n_in; (void)out_size; (void)ws_size;
  const float* x       = (const float*)d_in[0];
  const float* ln1_g   = (const float*)d_in[1];
  const float* ln1_b   = (const float*)d_in[2];
  const float* w_qkv   = (const float*)d_in[3];
  const float* b_qkv   = (const float*)d_in[4];
  const float* w_dense = (const float*)d_in[5];
  const float* b_dense = (const float*)d_in[6];
  const float* ln2_g   = (const float*)d_in[7];
  const float* ln2_b   = (const float*)d_in[8];
  const float* w_fc1   = (const float*)d_in[9];
  const float* w_fc2   = (const float*)d_in[10];
  const float* b_fc2   = (const float*)d_in[11];
  float* out = (float*)d_out;

  char* ws = (char*)d_ws;
  size_t off = 0;
  auto walloc = [&](size_t bytes) -> void* {
    void* p = ws + off;
    off += (bytes + 255) & ~(size_t)255;
    return p;
  };
  bf16*  wqkvT = (bf16*)walloc((size_t)3072 * 1024 * 2);
  bf16*  wdenT = (bf16*)walloc((size_t)1024 * 1024 * 2);
  bf16*  wfc1T = (bf16*)walloc((size_t)2048 * 1024 * 2);
  bf16*  wfc2T = (bf16*)walloc((size_t)1024 * 2048 * 2);
  bf16*  xn    = (bf16*)walloc((size_t)Mrows * Hh * 2);  // LN1 out; reused for LN2 out
  bf16*  qb    = (bf16*)walloc((size_t)Mrows * Hh * 2);
  bf16*  kb    = (bf16*)walloc((size_t)Mrows * Hh * 2);
  bf16*  vb    = (bf16*)walloc((size_t)Mrows * Hh * 2);
  bf16*  attnO = (bf16*)walloc((size_t)Mrows * Hh * 2);
  float* res1  = (float*)walloc((size_t)Mrows * Hh * 4);
  bf16*  h1    = qb;  // q+k regions (16 MB contiguous) reused for FC1 output

  const dim3 blk(256);

  // one-time weight transpose + bf16 conversion
  wconv_kernel<<<dim3((3072u * 1024u + 255u) / 256u), blk, 0, stream>>>(w_qkv,  wqkvT, 1024, 3072);
  wconv_kernel<<<dim3((1024u * 1024u + 255u) / 256u), blk, 0, stream>>>(w_dense, wdenT, 1024, 1024);
  wconv_kernel<<<dim3((1024u * 2048u + 255u) / 256u), blk, 0, stream>>>(w_fc1,  wfc1T, 1024, 2048);
  wconv_kernel<<<dim3((2048u * 1024u + 255u) / 256u), blk, 0, stream>>>(w_fc2,  wfc2T, 2048, 1024);

  // LN1
  ln_kernel<<<dim3(Mrows / 8), blk, 0, stream>>>(x, ln1_g, ln1_b, xn, Mrows);
  // QKV projection, scatter into per-head layout
  gemm_kernel<0><<<dim3(3072 / BN, Mrows / BM), blk, 0, stream>>>(
      xn, wqkvT, b_qkv, nullptr, nullptr, nullptr, qb, kb, vb, Mrows, 3072, 1024);
  // causal flash attention
  attn_kernel<<<dim3(Ss / QT, NH, Bb), blk, 0, stream>>>(qb, kb, vb, attnO);
  // dense proj + bias + residual(x) -> res1 (fp32)
  gemm_kernel<1><<<dim3(1024 / BN, Mrows / BM), blk, 0, stream>>>(
      attnO, wdenT, b_dense, x, res1, nullptr, nullptr, nullptr, nullptr,
      Mrows, 1024, 1024);
  // LN2 -> bf16 (reuse xn)
  ln_kernel<<<dim3(Mrows / 8), blk, 0, stream>>>(res1, ln2_g, ln2_b, xn, Mrows);
  // FC1 + exact GELU -> h1 (bf16)
  gemm_kernel<2><<<dim3(2048 / BN, Mrows / BM), blk, 0, stream>>>(
      xn, wfc1T, nullptr, nullptr, nullptr, h1, nullptr, nullptr, nullptr,
      Mrows, 2048, 1024);
  // FC2 + bias + residual(res1) -> d_out (fp32)
  gemm_kernel<1><<<dim3(1024 / BN, Mrows / BM), blk, 0, stream>>>(
      h1, wfc2T, b_fc2, res1, out, nullptr, nullptr, nullptr, nullptr,
      Mrows, 1024, 2048);
}
